// LSTMAutoencoder_18966575579317
// MI455X (gfx1250) — matrix operand order repaired and checked by reading, not compiled, not run on hardware
//
#include <hip/hip_runtime.h>
#include <hip/hip_bf16.h>

// ---------------------------------------------------------------------------
// LSTM autoencoder (72->64->32->64->72), B=512, T=512, CDNA5 (gfx1250) wave32.
//   1) cvt kernels: f32 weights -> zero-padded f16 ([4Hp][Kpad]); bias sum f32.
//   2) lstm_xgemm<DINP,N4P>: xg = x @ Wih^T + bias with v_wmma_f32_16x16x32_f16;
//      A fragments register-resident, B fragments double-buffered across N-tiles.
//   3) lstm_recurrent<HP,HPK>: 32 persistent blocks (16 batch rows each);
//      Whh staged into LDS via the Tensor Data Mover (tensor_load_to_lds +
//      s_wait_tensorcnt), B-fragments hoisted to registers for the whole T
//      loop, cell state in registers, xg software-pipelined one step ahead.
// ---------------------------------------------------------------------------

typedef __attribute__((ext_vector_type(16))) _Float16 v16h;
typedef __attribute__((ext_vector_type(8)))  float    v8f;
typedef __attribute__((ext_vector_type(4)))  unsigned int u32x4;
typedef __attribute__((ext_vector_type(8)))  int      i32x8;
typedef __attribute__((ext_vector_type(4)))  int      i32x4;

union Frag { v16h v; uint4 q[2]; _Float16 h[16]; };

#define BATCH 512
#define SEQT  512
#define MTOT  (BATCH * SEQT)      // 262144
#define MTILES (MTOT / 16)        // 16384
#define GEMM_WAVES 4

#if defined(__has_builtin)
#  if __has_builtin(__builtin_amdgcn_tensor_load_to_lds) && \
      __has_builtin(__builtin_amdgcn_s_wait_tensorcnt)
#    define USE_TDM 1
#  endif
#endif
#ifndef USE_TDM
#  define USE_TDM 0
#endif

__device__ __forceinline__ float fast_sigmoid(float x) {
  x = fminf(30.0f, fmaxf(-30.0f, x));
  return __builtin_amdgcn_rcpf(1.0f + __expf(-x));
}
__device__ __forceinline__ float fast_tanh(float x) {
  x = fminf(15.0f, fmaxf(-15.0f, x));
  const float e = __expf(2.0f * x);
  return (e - 1.0f) * __builtin_amdgcn_rcpf(e + 1.0f);
}

// ---------------------------------------------------------------------------
// Weight conversion: f32 [4H][Din] -> f16 [4Hp][DinP], gate-segment padded.
// Also used for Whh with Din=H, DinP=HpK.
// ---------------------------------------------------------------------------
__global__ void cvt_w_f16(const float* __restrict__ W, _Float16* __restrict__ Wo,
                          int H, int Hp, int Din, int DinP) {
  int idx = blockIdx.x * 256 + threadIdx.x;
  int total = 4 * Hp * DinP;
  if (idx >= total) return;
  int k  = idx % DinP;
  int np = idx / DinP;
  int gate = np / Hp, wi = np % Hp;
  float v = (wi < H && k < Din) ? W[(size_t)(gate * H + wi) * Din + k] : 0.0f;
  Wo[idx] = (_Float16)v;
}

__global__ void cvt_bias(const float* __restrict__ bih, const float* __restrict__ bhh,
                         float* __restrict__ bo, int H, int Hp) {
  int idx = blockIdx.x * 256 + threadIdx.x;
  if (idx >= 4 * Hp) return;
  int gate = idx / Hp, wi = idx % Hp;
  bo[idx] = (wi < H) ? (bih[gate * H + wi] + bhh[gate * H + wi]) : 0.0f;
}

// ---------------------------------------------------------------------------
// Input GEMM: xg[M][N4P] = inAct[M][Din] @ W^T + bias   (W is [4Hp][DINP] f16)
// A fragments (activations) in registers; B double-buffered across N-tiles.
// A-frag per lane: two 16B chunks at K = kk + 8*(lane>=16) and +16 (row lane%16).
// B-frag per lane: two 16B chunks at K = kk + 8*(lane>=16) and +16 of row n = nbase + lane%16 of W.
// C/D element: (M = v + 8*(lane>=16), N = lane%16).
// ---------------------------------------------------------------------------
template <bool F32IN, int DINP, int N4P>
__global__ void lstm_xgemm(const void* __restrict__ inAct, int inStride, int Din,
                           const _Float16* __restrict__ W, const float* __restrict__ bias,
                           _Float16* __restrict__ xg) {
  constexpr int KT = DINP / 32;
  constexpr int NT = N4P / 16;
  const int lane = threadIdx.x & 31;
  const int wave = threadIdx.x >> 5;
  const int llo  = lane & 15;
  const int lhi  = lane >> 4;
  const int mtile = blockIdx.x * GEMM_WAVES + wave;
  const int rowA  = mtile * 16 + llo;

  // ---- load all A fragments once ---------------------------------------
  Frag a[KT];
  #pragma unroll
  for (int i = 0; i < KT; ++i) {
    const int c0 = i * 32 + 8 * lhi;   // halves 0..7
    const int c1 = c0 + 16;            // halves 8..15
    if (F32IN) {
      const float* xr = (const float*)inAct + (size_t)rowA * inStride;
      #pragma unroll
      for (int j = 0; j < 8; ++j) a[i].h[j]     = (c0 < Din) ? (_Float16)xr[c0 + j] : (_Float16)0.0f;
      #pragma unroll
      for (int j = 0; j < 8; ++j) a[i].h[8 + j] = (c1 < Din) ? (_Float16)xr[c1 + j] : (_Float16)0.0f;
    } else {
      const _Float16* xr = (const _Float16*)inAct + (size_t)rowA * inStride;
      a[i].q[0] = *(const uint4*)(xr + c0);
      a[i].q[1] = *(const uint4*)(xr + c1);
    }
  }

  // ---- sweep N-tiles, reusing A; double-buffer B ------------------------
  Frag b[2][KT];
  #pragma unroll
  for (int i = 0; i < KT; ++i) {
    const _Float16* wr = W + (size_t)llo * DINP + i * 32 + 8 * lhi;
    b[0][i].q[0] = *(const uint4*)(wr);
    b[0][i].q[1] = *(const uint4*)(wr + 16);
  }
  #pragma unroll
  for (int nt = 0; nt < NT; ++nt) {
    if (nt + 1 < NT) {
      #pragma unroll
      for (int i = 0; i < KT; ++i) {
        const _Float16* wr = W + (size_t)((nt + 1) * 16 + llo) * DINP + i * 32 + 8 * lhi;
        b[(nt + 1) & 1][i].q[0] = *(const uint4*)(wr);
        b[(nt + 1) & 1][i].q[1] = *(const uint4*)(wr + 16);
      }
    }
    v8f acc = {};
    #pragma unroll
    for (int i = 0; i < KT; ++i)
      acc = __builtin_amdgcn_wmma_f32_16x16x32_f16(false, a[i].v, false, b[nt & 1][i].v,
                                                   (short)0, acc, false, false);
    const int nbase = nt * 16;
    const float bv = bias[nbase + llo];
    #pragma unroll
    for (int v = 0; v < 8; ++v) {
      const int m = mtile * 16 + v + 8 * lhi;
      xg[(size_t)m * N4P + nbase + llo] = (_Float16)(acc[v] + bv);
    }
  }
}

// ---------------------------------------------------------------------------
// Recurrent scan. grid = B/16 blocks; blockDim = (4*HP/16) waves * 32.
// Per timestep: acc = xg(t) [pipelined]; acc += h@Whh^T (WMMA, B in regs);
// per-gate activation -> sG; distributed combine (2 cells/thread, c in regs).
// Whh staged into LDS with the Tensor Data Mover when available.
// ---------------------------------------------------------------------------
template <int HP, int HPK, bool FINAL>
__global__ void lstm_recurrent(const _Float16* __restrict__ xg,   // [M][N4P] f16
                               const _Float16* __restrict__ Whh,  // [4HP][HPK] f16
                               int H, int T,
                               _Float16* __restrict__ actOut,     // [M][HP] f16
                               float* __restrict__ outF32) {      // [M][H] f32
  constexpr int N4P = 4 * HP;
  constexpr int KT  = HPK / 32;
  constexpr int nIT = HP / 16;

  __shared__ _Float16 sWhh[N4P * HPK];
  __shared__ _Float16 sH[16 * HPK];     // h (f16), K-padded, pad cols stay 0
  __shared__ float    sG[16 * N4P];     // activated gates

  const int tid  = threadIdx.x;
  const int lane = tid & 31;
  const int wave = tid >> 5;
  const int llo  = lane & 15;
  const int lhi  = lane >> 4;
  const int nthreads = blockDim.x;
  const int btile = blockIdx.x;
  const int nbase = wave * 16;
  const int gate  = wave / nIT;         // 0:i 1:f 2:g 3:o

#if USE_TDM
  // ---- TDM: DMA Whh ([N4P][HPK] f16, fully dense 2-D tile) into LDS -----
  if (wave == 0) {
    const unsigned lds_addr = (unsigned)(unsigned long long)(const void*)sWhh;
    const unsigned long long ga = (unsigned long long)(const void*)Whh;
    u32x4 g0;
    g0[0] = 1u;                                              // count=1 (valid user D#)
    g0[1] = lds_addr;                                        // lds_addr (bytes)
    g0[2] = (unsigned)(ga & 0xFFFFFFFFu);                    // global_addr[31:0]
    g0[3] = (unsigned)((ga >> 32) & 0x01FFFFFFu) | (2u << 30); // addr[56:32] | type=2
    i32x8 g1;
    g1[0] = (int)(1u << 16);                                 // wg_mask=0, data_size=1 (2B)
    g1[1] = (int)(((unsigned)HPK & 0xFFFFu) << 16);          // tensor_dim0.lo16
    g1[2] = (int)((((unsigned)HPK >> 16) & 0xFFFFu) |        // tensor_dim0.hi16
                  (((unsigned)N4P & 0xFFFFu) << 16));        // tensor_dim1.lo16
    g1[3] = (int)((((unsigned)N4P >> 16) & 0xFFFFu) |        // tensor_dim1.hi16
                  (((unsigned)HPK & 0xFFFFu) << 16));        // tile_dim0 = HPK
    g1[4] = (int)((unsigned)N4P & 0xFFFFu);                  // tile_dim1 = N4P, tile_dim2=0
    g1[5] = (int)(unsigned)HPK;                              // tensor_dim0_stride.lo32
    g1[6] = 0;                                               // stride0.hi16 | stride1.lo16
    g1[7] = 0;                                               // stride1 rest
    i32x4 z4 = {0, 0, 0, 0};
#if defined(__clang_major__) && (__clang_major__ >= 23)
    i32x8 z8 = {0, 0, 0, 0, 0, 0, 0, 0};
    __builtin_amdgcn_tensor_load_to_lds(g0, g1, z4, z4, z8, 0);
#else
    __builtin_amdgcn_tensor_load_to_lds(g0, g1, z4, z4, 0);
#endif
    __builtin_amdgcn_s_wait_tensorcnt(0);
  }
#else
  for (int i = tid; i < N4P * HPK; i += nthreads) sWhh[i] = Whh[i];
#endif
  for (int i = tid; i < 16 * HPK; i += nthreads) sH[i] = (_Float16)0.0f;
  __syncthreads();

  // ---- B fragments (Whh) are invariant over time: hoist to registers ----
  Frag bf[KT];
  #pragma unroll
  for (int i = 0; i < KT; ++i) {
    const _Float16* wr = sWhh + (size_t)(nbase + llo) * HPK + i * 32 + 8 * lhi;
    bf[i].q[0] = *(const uint4*)(wr);
    bf[i].q[1] = *(const uint4*)(wr + 16);
  }

  // ---- distributed combine ownership: 2 adjacent cells per thread -------
  const int eidx = tid * 2;             // flat over [16][HP]
  const int cm = eidx / HP;             // row (batch within tile)
  const int cc = eidx % HP;             // column pair base (even)
  float creg0 = 0.0f, creg1 = 0.0f;     // cell state lives in registers

  // ---- software-pipelined xg loads --------------------------------------
  _Float16 xv[8];
  #pragma unroll
  for (int v = 0; v < 8; ++v)
    xv[v] = xg[(size_t)(btile * 16 + v + 8 * lhi) * T * N4P + nbase + llo];

  for (int t = 0; t < T; ++t) {
    v8f acc;
    #pragma unroll
    for (int v = 0; v < 8; ++v) acc[v] = (float)xv[v];
    if (t + 1 < T) {
      #pragma unroll
      for (int v = 0; v < 8; ++v)
        xv[v] = xg[((size_t)(btile * 16 + v + 8 * lhi) * T + (t + 1)) * N4P + nbase + llo];
    }

    // ---- recurrent WMMA: all A loads first, then back-to-back WMMAs ----
    Frag af[KT];
    #pragma unroll
    for (int i = 0; i < KT; ++i) {
      const _Float16* hr = sH + llo * HPK + i * 32;
      af[i].q[0] = *(const uint4*)(hr + 8 * lhi);
      af[i].q[1] = *(const uint4*)(hr + 16 + 8 * lhi);
    }
    #pragma unroll
    for (int i = 0; i < KT; ++i)
      acc = __builtin_amdgcn_wmma_f32_16x16x32_f16(false, af[i].v, false, bf[i].v,
                                                   (short)0, acc, false, false);

    // ---- per-gate activation into LDS ----------------------------------
    #pragma unroll
    for (int v = 0; v < 8; ++v) {
      const float g = acc[v];
      const float aout = (gate == 2) ? fast_tanh(g) : fast_sigmoid(g);
      sG[(v + 8 * lhi) * N4P + nbase + llo] = aout;
    }
    __syncthreads();

    // ---- combine: every thread updates its 2 cells ---------------------
    {
      const float2 iv = *(const float2*)&sG[cm * N4P + cc];
      const float2 fv = *(const float2*)&sG[cm * N4P + HP + cc];
      const float2 gv = *(const float2*)&sG[cm * N4P + 2 * HP + cc];
      const float2 ov = *(const float2*)&sG[cm * N4P + 3 * HP + cc];
      creg0 = fv.x * creg0 + iv.x * gv.x;
      creg1 = fv.y * creg1 + iv.y * gv.y;
      const float h0 = ov.x * fast_tanh(creg0);
      const float h1 = ov.y * fast_tanh(creg1);

      union { _Float16 h[2]; unsigned u; } hp;
      hp.h[0] = (_Float16)h0;
      hp.h[1] = (_Float16)h1;
      *(unsigned*)&sH[cm * HPK + cc] = hp.u;

      const size_t M = (size_t)(btile * 16 + cm) * T + t;
      if (FINAL) {
        if (cc < H) {
          float2 o2; o2.x = h0; o2.y = h1;
          *(float2*)&outF32[M * H + cc] = o2;
        }
      } else {
        *(unsigned*)&actOut[M * HP + cc] = hp.u;
      }
    }
    __syncthreads();
  }
}

// ---------------------------------------------------------------------------
// Host orchestration
// ---------------------------------------------------------------------------
struct LayerCfg { int Din, DinP, H, Hp, HpK, N4p; };

extern "C" void kernel_launch(void* const* d_in, const int* in_sizes, int n_in,
                              void* d_out, int out_size, void* d_ws, size_t ws_size,
                              hipStream_t stream) {
  (void)in_sizes; (void)n_in; (void)out_size; (void)ws_size;

  static const LayerCfg cfg[4] = {
    {72, 96, 64, 64, 64, 256},   // e1: 72 -> 64
    {64, 64, 32, 32, 32, 128},   // e2: 64 -> 32
    {32, 32, 64, 64, 64, 256},   // d1: 32 -> 64
    {64, 64, 72, 80, 96, 320},   // d2: 64 -> 72 (padded to 80/96)
  };

  const size_t XG_BYTES  = (size_t)MTOT * 320 * sizeof(_Float16);  // 160 MB
  const size_t ACT_BYTES = (size_t)MTOT * 64 * sizeof(_Float16);   // 32 MB
  char* ws = (char*)d_ws;
  _Float16* xg    = (_Float16*)(ws);
  _Float16* actA  = (_Float16*)(ws + XG_BYTES);
  _Float16* actB  = (_Float16*)(ws + XG_BYTES + ACT_BYTES);
  _Float16* wih16 = (_Float16*)(ws + XG_BYTES + 2 * ACT_BYTES);
  _Float16* whh16 = (_Float16*)(ws + XG_BYTES + 2 * ACT_BYTES + 65536);
  float*    biasb = (float*)   (ws + XG_BYTES + 2 * ACT_BYTES + 2 * 65536);

  const float* x = (const float*)d_in[0];
  float* out = (float*)d_out;
  const dim3 ggrid(MTILES / GEMM_WAVES);

  for (int L = 0; L < 4; ++L) {
    const LayerCfg c = cfg[L];
    const float* Wih = (const float*)d_in[1 + 4 * L + 0];
    const float* Whh = (const float*)d_in[1 + 4 * L + 1];
    const float* bih = (const float*)d_in[1 + 4 * L + 2];
    const float* bhh = (const float*)d_in[1 + 4 * L + 3];

    int tw = 4 * c.Hp * c.DinP;
    cvt_w_f16<<<(tw + 255) / 256, 256, 0, stream>>>(Wih, wih16, c.H, c.Hp, c.Din, c.DinP);
    tw = 4 * c.Hp * c.HpK;
    cvt_w_f16<<<(tw + 255) / 256, 256, 0, stream>>>(Whh, whh16, c.H, c.Hp, c.H, c.HpK);
    tw = 4 * c.Hp;
    cvt_bias<<<(tw + 255) / 256, 256, 0, stream>>>(bih, bhh, biasb, c.H, c.Hp);

    switch (L) {
      case 0:
        lstm_xgemm<true, 96, 256><<<ggrid, GEMM_WAVES * 32, 0, stream>>>(
            x, 72, 72, wih16, biasb, xg);
        lstm_recurrent<64, 64, false><<<BATCH / 16, 512, 0, stream>>>(
            xg, whh16, 64, SEQT, actA, nullptr);
        break;
      case 1:
        lstm_xgemm<false, 64, 128><<<ggrid, GEMM_WAVES * 32, 0, stream>>>(
            actA, 64, 64, wih16, biasb, xg);
        lstm_recurrent<32, 32, false><<<BATCH / 16, 256, 0, stream>>>(
            xg, whh16, 32, SEQT, actB, nullptr);
        break;
      case 2:
        lstm_xgemm<false, 32, 256><<<ggrid, GEMM_WAVES * 32, 0, stream>>>(
            actB, 32, 32, wih16, biasb, xg);
        lstm_recurrent<64, 64, false><<<BATCH / 16, 512, 0, stream>>>(
            xg, whh16, 64, SEQT, actA, nullptr);
        break;
      case 3:
        lstm_xgemm<false, 64, 320><<<ggrid, GEMM_WAVES * 32, 0, stream>>>(
            actA, 64, 64, wih16, biasb, xg);
        lstm_recurrent<80, 96, true><<<BATCH / 16, 640, 0, stream>>>(
            xg, whh16, 72, SEQT, nullptr, out);
        break;
    }
  }
}
